// AttentionXL_22204980920325
// MI455X (gfx1250) — matrix operand order repaired and checked by reading, not compiled, not run on hardware
//
#include <hip/hip_runtime.h>

// ---------------------------------------------------------------------------
// TransformerXL relative attention for MI455X (gfx1250, wave32, WMMA bf16,
// async global->LDS DMA with static ping-pong double-buffering).
// Pipeline:
//   0) pack inputs/pos/full to bf16; pack weights to bf16 TRANSPOSED (NxK)
//   1) kv  = full_input @ W_kv + b_kv   -> bf16      (4096 x 1024 x 2048)
//   2) q   = inputs     @ W_q  + b_q    -> f32       (2048 x 1024 x 1024)
//   3) r   = pos_emb    @ W_pos+ b_pos  -> bf16      (1024 x 1024 x 1024)
//   4) fused flash attention w/ rel-shift -> bf16 attn_vec
//   5) out = attn_vec   @ W_proj + b_proj -> f32 d_out
// GEMM: 128x128 tile, K=1024 (KT=32 steps of 32), LDS filled with
// global_load_async_to_lds_b128 (ASYNCcnt), fragments via ds_load_b128,
// compute via v_wmma_f32_16x16x32_bf16. Two K-steps unrolled per loop
// iteration so buffer indices / waits are compile-time static.
// ---------------------------------------------------------------------------

#define CUR  512
#define FULL 1024
#define BSZ  4
#define DIMM 1024
#define NH   16
#define HD   64
#define PREV (FULL - CUR)
#define KDIM 1024            // shared K (and row stride) of every GEMM here
#define KT   (KDIM / 32)     // 32 K-steps

typedef __attribute__((ext_vector_type(16))) __bf16 v16bf;
typedef __attribute__((ext_vector_type(8)))  float  v8f;

__device__ static inline __bf16 f2bf(float f) {
  unsigned u = __builtin_bit_cast(unsigned, f);
  u += 0x7fffu + ((u >> 16) & 1u);               // round-to-nearest-even
  unsigned short hs = (unsigned short)(u >> 16);
  return __builtin_bit_cast(__bf16, hs);
}

// Generic LDS pointer: its low 32 bits are the LDS offset.
__device__ static inline uint32_t lds_off(const void* p) {
  return (uint32_t)(uintptr_t)p;
}

// Async DMA: copy 16 bytes / lane from global to LDS (tracked by ASYNCcnt).
__device__ static inline void async_b128(uint32_t lds, const void* gaddr) {
  asm volatile("global_load_async_to_lds_b128 %0, %1, off"
               :: "v"(lds), "v"(gaddr) : "memory");
}
#define ASYNC_WAIT_4() asm volatile("s_wait_asynccnt 0x4" ::: "memory")
#define ASYNC_WAIT_0() asm volatile("s_wait_asynccnt 0x0" ::: "memory")

// ---------------------------------------------------------------------------
// f32 -> bf16 flat pack (n multiple of 4)
// ---------------------------------------------------------------------------
__global__ __launch_bounds__(256) void pack_bf16_kernel(
    const float* __restrict__ src, __bf16* __restrict__ dst, int n) {
  const int idx = (blockIdx.x * 256 + threadIdx.x) * 4;
  if (idx < n) {
#pragma unroll
    for (int j = 0; j < 4; ++j) dst[idx + j] = f2bf(src[idx + j]);
  }
}

// f32 (K x N) -> bf16 transposed (N x K): dst[n*K+k] = src[k*N+n]
__global__ __launch_bounds__(256) void pack_bf16_t_kernel(
    const float* __restrict__ src, __bf16* __restrict__ dst, int K, int N) {
  const int idx = blockIdx.x * 256 + threadIdx.x;   // over N*K, writes coalesced
  const int n = idx / K, k = idx - n * K;
  if (n < N) dst[(size_t)n * K + k] = f2bf(src[(size_t)k * N + n]);
}

// ---------------------------------------------------------------------------
// GEMM: C = A[Mx1024](bf16,row) @ Bt[Nx1024](bf16,row)^T + bias.
// Block 256 (8 waves); wave (wm 0..3, wn 0..1) owns 32x64 = 2x4 frags.
// ---------------------------------------------------------------------------
__global__ __launch_bounds__(256) void gemm_bf16_kernel(
    const __bf16* __restrict__ A, const __bf16* __restrict__ Bt,
    const float* __restrict__ bias, float* __restrict__ Cf,
    __bf16* __restrict__ Cb, int M, int N)
{
  __shared__ __align__(16) __bf16 Ab[2][128 * 32];
  __shared__ __align__(16) __bf16 Bb[2][128 * 32];   // [n][k] (Bt tile verbatim)

  const int tid  = threadIdx.x;
  const int lane = tid & 31;
  const int gg   = lane >> 4;
  const int ln   = lane & 15;
  const int wid  = tid >> 5;
  const int wm   = wid >> 1;
  const int wn   = wid & 1;
  const int m0   = blockIdx.y * 128;
  const int n0   = blockIdx.x * 128;

  // Per-thread async chunks: tile = 128 rows x 64B; chunk c (0..511) -> 16B.
  const int c0 = tid, c1 = tid + 256;
  const __bf16* gA0 = A  + (size_t)(m0 + (c0 >> 2)) * KDIM + (c0 & 3) * 8;
  const __bf16* gA1 = A  + (size_t)(m0 + (c1 >> 2)) * KDIM + (c1 & 3) * 8;
  const __bf16* gB0 = Bt + (size_t)(n0 + (c0 >> 2)) * KDIM + (c0 & 3) * 8;
  const __bf16* gB1 = Bt + (size_t)(n0 + (c1 >> 2)) * KDIM + (c1 & 3) * 8;
  const uint32_t lA0 = lds_off(&Ab[0][0]) + c0 * 16;
  const uint32_t lA1 = lds_off(&Ab[0][0]) + c1 * 16;
  const uint32_t lB0 = lds_off(&Bb[0][0]) + c0 * 16;
  const uint32_t lB1 = lds_off(&Bb[0][0]) + c1 * 16;
  const uint32_t ldsBufStride = 128 * 32 * 2;       // bytes between buf0/buf1

  v8f acc[2][4] = {};

  // issue K-step `step` into buffer `buf` (both compile-time-friendly)
  auto issue = [&](uint32_t bufOff, int step) {
    const int ko = step * 32;
    async_b128(lA0 + bufOff, gA0 + ko);
    async_b128(lA1 + bufOff, gA1 + ko);
    async_b128(lB0 + bufOff, gB0 + ko);
    async_b128(lB1 + bufOff, gB1 + ko);
  };
  // consume buffer `buf`: build fragments, 8 WMMAs
  auto compute = [&](const __bf16* __restrict__ abuf,
                     const __bf16* __restrict__ bbuf) {
    v16bf af[2], bfr[4];
#pragma unroll
    for (int fm = 0; fm < 2; ++fm) {
      const int m = wm * 32 + fm * 16 + ln;
#pragma unroll
      for (int e = 0; e < 16; ++e) {
        const int kk = gg * 16 + e;
        af[fm][e] = abuf[m * 32 + kk];
      }
    }
#pragma unroll
    for (int fn = 0; fn < 4; ++fn) {
      const int n = wn * 64 + fn * 16 + ln;
#pragma unroll
      for (int e = 0; e < 16; ++e) bfr[fn][e] = bbuf[n * 32 + gg * 16 + e];
    }
#pragma unroll
    for (int fm = 0; fm < 2; ++fm)
#pragma unroll
      for (int fn = 0; fn < 4; ++fn)
        acc[fm][fn] = __builtin_amdgcn_wmma_f32_16x16x32_bf16(
            false, af[fm], false, bfr[fn], (short)0, acc[fm][fn], false, false);
  };

  issue(0, 0);              // step 0 -> buf0
  issue(ldsBufStride, 1);   // step 1 -> buf1

  for (int kt = 0; kt < KT; kt += 2) {
    const bool more = (kt + 2) < KT;    // same condition for kt+3 (kt even)

    ASYNC_WAIT_4();                     // step kt landed (kt+1 in flight)
    __syncthreads();
    compute(&Ab[0][0], &Bb[0][0]);
    __syncthreads();
    if (more) issue(0, kt + 2);         // refill buf0

    if (more) { ASYNC_WAIT_4(); } else { ASYNC_WAIT_0(); }  // step kt+1 landed
    __syncthreads();
    compute(&Ab[1][0], &Bb[1][0]);
    __syncthreads();
    if (more) issue(ldsBufStride, kt + 3);                  // refill buf1
  }

#pragma unroll
  for (int fm = 0; fm < 2; ++fm)
#pragma unroll
    for (int fn = 0; fn < 4; ++fn)
#pragma unroll
      for (int g = 0; g < 8; ++g) {
        const int m = m0 + wm * 32 + fm * 16 + gg * 8 + g;
        const int n = n0 + wn * 64 + fn * 16 + ln;
        const float val = acc[fm][fn][g] + bias[n];
        if (Cf) Cf[(size_t)m * N + n] = val;
        if (Cb) Cb[(size_t)m * N + n] = f2bf(val);
      }
}

// ---------------------------------------------------------------------------
// Fused flash attention with TransformerXL rel-shift (one wave per 16-row
// query tile of one (b,h)). K/V/R consumed as bf16 (no conversions).
// ---------------------------------------------------------------------------
__global__ __launch_bounds__(32) void attn_xl_kernel(
    const float*  __restrict__ q,   // (CUR, BSZ, NH, HD) f32
    const __bf16* __restrict__ kv,  // (FULL, BSZ, 2*NH*HD) bf16  key | value
    const __bf16* __restrict__ r,   // (FULL, NH, HD) bf16
    const float*  __restrict__ u,   // (NH, HD)
    const float*  __restrict__ v,   // (NH, HD)
    __bf16* __restrict__ av)        // (CUR, BSZ, NH, HD) bf16
{
  const int lane = threadIdx.x;
  const int gg   = lane >> 4;
  const int ln   = lane & 15;
  const int i0   = blockIdx.x * 16;
  const int b    = blockIdx.y / NH;
  const int h    = blockIdx.y % NH;

  __shared__ __align__(16) __bf16 Plds[16][32];

  v16bf aqu[2], aqv[2];
  {
    const int i = i0 + ln;
    const float* qrow = q + (((size_t)i * BSZ + b) * NH + h) * HD;
    const float* urow = u + h * HD;
    const float* vrow = v + h * HD;
#pragma unroll
    for (int ck = 0; ck < 2; ++ck)
#pragma unroll
      for (int e = 0; e < 16; ++e) {
        const int d = ck * 32 + gg * 16 + e;
        const float qd = qrow[d];
        aqu[ck][e] = f2bf(qd + urow[d]);
        aqv[ck][e] = f2bf(qd + vrow[d]);
      }
  }

  v8f o[4] = {};
  float mrow[8], lrow[8];
#pragma unroll
  for (int g = 0; g < 8; ++g) { mrow[g] = -1e30f; lrow[g] = 0.f; }

  const size_t kvStride = (size_t)BSZ * 2 * NH * HD;
  const __bf16* keyBase = kv + (size_t)b * (2 * NH * HD) + h * HD;
  const __bf16* valBase = keyBase + NH * HD;

  const int iters = ((i0 + 15 + PREV) >> 5) + 1;
  for (int it = 0; it < iters; ++it) {
    const int j0 = it * 32;

    // content scores: (q+u) . k
    v8f s[2];
#pragma unroll
    for (int t = 0; t < 2; ++t) {
      v8f a = {};
      const int j = j0 + 16 * t + ln;
#pragma unroll
      for (int ck = 0; ck < 2; ++ck) {
        const __bf16* krow = keyBase + (size_t)j * kvStride + ck * 32 + gg * 16;
        v16bf bk;
#pragma unroll
        for (int e = 0; e < 16; ++e) bk[e] = krow[e];
        a = __builtin_amdgcn_wmma_f32_16x16x32_bf16(false, aqu[ck], false, bk,
                                                    (short)0, a, false, false);
      }
      s[t] = a;
    }

    // raw positional scores: 3 tiles cover the shift window base..base+46
    const int base = j0 + (CUR - 16) - i0;
    v8f p[3];
#pragma unroll
    for (int tt = 0; tt < 3; ++tt) {
      v8f a = {};
      int kr = base + 16 * tt + ln;
      if (kr > FULL - 1) kr = FULL - 1;          // masked region only
      const __bf16* rrow = r + ((size_t)kr * NH + h) * HD;
#pragma unroll
      for (int ck = 0; ck < 2; ++ck) {
        v16bf br;
#pragma unroll
        for (int e = 0; e < 16; ++e) br[e] = rrow[ck * 32 + gg * 16 + e];
        a = __builtin_amdgcn_wmma_f32_16x16x32_bf16(false, aqv[ck], false, br,
                                                    (short)0, a, false, false);
      }
      p[tt] = a;
    }

    // rel-shift gather (cross-lane), combine, scale, causal mask
#pragma unroll
    for (int t = 0; t < 2; ++t)
#pragma unroll
      for (int g = 0; g < 8; ++g) {
        const int m   = gg * 8 + g;
        const int off = 16 * t + ln + 15 - m;
        const int src = gg * 16 + (off & 15);
        const float plo = __shfl(p[t][g],     src, 32);
        const float phi = __shfl(p[t + 1][g], src, 32);
        const float posv = ((off >> 4) == t) ? plo : phi;
        float sv = (s[t][g] + posv) * 0.125f;
        const int j = j0 + 16 * t + ln;
        if (j > i0 + m + PREV) sv = -1e30f;
        s[t][g] = sv;
      }

    // online softmax
    __syncthreads();
#pragma unroll
    for (int g = 0; g < 8; ++g) {
      float tmax = fmaxf(s[0][g], s[1][g]);
#pragma unroll
      for (int w = 1; w < 16; w <<= 1) tmax = fmaxf(tmax, __shfl_xor(tmax, w, 32));
      const float mnew  = fmaxf(mrow[g], tmax);
      const float alpha = __expf(mrow[g] - mnew);
      const float e0 = __expf(s[0][g] - mnew);
      const float e1 = __expf(s[1][g] - mnew);
      float rs = e0 + e1;
#pragma unroll
      for (int w = 1; w < 16; w <<= 1) rs += __shfl_xor(rs, w, 32);
      lrow[g] = lrow[g] * alpha + rs;
      mrow[g] = mnew;
#pragma unroll
      for (int c = 0; c < 4; ++c) o[c][g] *= alpha;
      const int m = gg * 8 + g;
      Plds[m][ln]      = f2bf(e0);
      Plds[m][16 + ln] = f2bf(e1);
    }
    __syncthreads();

    v16bf ap;
#pragma unroll
    for (int e = 0; e < 16; ++e) {
      const int k = gg * 16 + e;
      ap[e] = Plds[ln][k];
    }

    // O += P x V
#pragma unroll
    for (int c = 0; c < 4; ++c) {
      v16bf bv;
#pragma unroll
      for (int e = 0; e < 16; ++e) {
        const int j = j0 + gg * 16 + e;
        bv[e] = valBase[(size_t)j * kvStride + c * 16 + ln];
      }
      o[c] = __builtin_amdgcn_wmma_f32_16x16x32_bf16(false, ap, false, bv,
                                                     (short)0, o[c], false, false);
    }
  }

#pragma unroll
  for (int g = 0; g < 8; ++g) {
    const int m = gg * 8 + g;
    const float inv = 1.f / lrow[g];
#pragma unroll
    for (int c = 0; c < 4; ++c) {
      const int d = c * 16 + ln;
      av[(((size_t)(i0 + m) * BSZ + b) * NH + h) * HD + d] = f2bf(o[c][g] * inv);
    }
  }
}

// ---------------------------------------------------------------------------
extern "C" void kernel_launch(void* const* d_in, const int* in_sizes, int n_in,
                              void* d_out, int out_size, void* d_ws, size_t ws_size,
                              hipStream_t stream) {
  (void)in_sizes; (void)n_in; (void)out_size; (void)ws_size;
  const float* inputs    = (const float*)d_in[0];
  const float* pos_emb   = (const float*)d_in[1];
  const float* full_inp  = (const float*)d_in[2];
  const float* u         = (const float*)d_in[3];
  const float* v         = (const float*)d_in[4];
  const float* W_kv      = (const float*)d_in[5];
  const float* b_kv      = (const float*)d_in[6];
  const float* W_q       = (const float*)d_in[7];
  const float* b_q       = (const float*)d_in[8];
  const float* W_pos     = (const float*)d_in[9];
  const float* b_pos     = (const float*)d_in[10];
  const float* W_proj    = (const float*)d_in[11];
  const float* b_proj    = (const float*)d_in[12];
  float* out = (float*)d_out;

  // sizes (elements)
  const size_t nInp  = (size_t)CUR * BSZ * DIMM;          // 2M
  const size_t nFull = (size_t)FULL * BSZ * DIMM;         // 4M
  const size_t nPos  = (size_t)FULL * DIMM;               // 1M
  const size_t nWkv  = (size_t)DIMM * 2 * NH * HD;        // 2M
  const size_t nW    = (size_t)DIMM * NH * HD;            // 1M
  const size_t nKV   = (size_t)FULL * BSZ * 2 * NH * HD;  // 8.4M
  const size_t nQ    = (size_t)CUR * BSZ * NH * HD;       // 2M
  const size_t nR    = (size_t)FULL * NH * HD;            // 1M

  // workspace: f32 q | bf16 everything else
  float*  qf      = (float*)d_ws;
  __bf16* inp_bf  = (__bf16*)(qf + nQ);
  __bf16* full_bf = inp_bf  + nInp;
  __bf16* pos_bf  = full_bf + nFull;
  __bf16* wkv_t   = pos_bf  + nPos;
  __bf16* wq_t    = wkv_t   + nWkv;
  __bf16* wpos_t  = wq_t    + nW;
  __bf16* wproj_t = wpos_t  + nW;
  __bf16* kv_bf   = wproj_t + nW;
  __bf16* r_bf    = kv_bf   + nKV;
  __bf16* av_bf   = r_bf    + nR;

  // 0) packs
  pack_bf16_kernel<<<nInp  / 1024, 256, 0, stream>>>(inputs,   inp_bf,  (int)nInp);
  pack_bf16_kernel<<<nFull / 1024, 256, 0, stream>>>(full_inp, full_bf, (int)nFull);
  pack_bf16_kernel<<<nPos  / 1024, 256, 0, stream>>>(pos_emb,  pos_bf,  (int)nPos);
  pack_bf16_t_kernel<<<nWkv / 256, 256, 0, stream>>>(W_kv,   wkv_t,   DIMM, 2 * NH * HD);
  pack_bf16_t_kernel<<<nW   / 256, 256, 0, stream>>>(W_q,    wq_t,    DIMM, NH * HD);
  pack_bf16_t_kernel<<<nW   / 256, 256, 0, stream>>>(W_pos,  wpos_t,  DIMM, NH * HD);
  pack_bf16_t_kernel<<<nW   / 256, 256, 0, stream>>>(W_proj, wproj_t, NH * HD, DIMM);

  // 1) kv projection -> bf16
  gemm_bf16_kernel<<<dim3((2 * NH * HD) / 128, (FULL * BSZ) / 128), 256, 0, stream>>>(
      full_bf, wkv_t, b_kv, nullptr, kv_bf, FULL * BSZ, 2 * NH * HD);
  // 2) q projection -> f32
  gemm_bf16_kernel<<<dim3((NH * HD) / 128, (CUR * BSZ) / 128), 256, 0, stream>>>(
      inp_bf, wq_t, b_q, qf, nullptr, CUR * BSZ, NH * HD);
  // 3) pos projection -> bf16
  gemm_bf16_kernel<<<dim3((NH * HD) / 128, FULL / 128), 256, 0, stream>>>(
      pos_bf, wpos_t, b_pos, nullptr, r_bf, FULL, NH * HD);
  // 4) fused rel-shift flash attention -> bf16 attn_vec
  attn_xl_kernel<<<dim3(CUR / 16, BSZ * NH), 32, 0, stream>>>(
      qf, kv_bf, r_bf, u, v, av_bf);
  // 5) output projection -> f32 d_out
  gemm_bf16_kernel<<<dim3(DIMM / 128, (CUR * BSZ) / 128), 256, 0, stream>>>(
      av_bf, wproj_t, b_proj, out, nullptr, CUR * BSZ, DIMM);
}
